// Clustering_Layer_74938589381104
// MI455X (gfx1250) — matrix operand order repaired and checked by reading, not compiled, not run on hardware
//
#include <hip/hip_runtime.h>
#include <hip/hip_bf16.h>

typedef __attribute__((ext_vector_type(16))) __bf16 v16bf;
typedef __attribute__((ext_vector_type(8)))  __bf16 v8bf;
typedef __attribute__((ext_vector_type(8)))  float  v8f;
typedef __attribute__((ext_vector_type(4)))  float  v4f;

#define KCLUST   100
#define DDIM     256
#define NTILES   7                      // ceil(100/16) cluster tiles
#define KSTEPS   8                      // 256 / 32
#define TSTEPS   (NTILES * KSTEPS)      // 56 (tile,kstep) pairs
#define ROWT     2                      // row tiles per wave -> 32 rows/wave

// Packed B layout, identical in global ws and LDS, in 16-byte records.
// Each WMMA B operand (32B/lane) is split into two 16B chunks stored
// chunk-major so the per-lane stride is 16B -> conflict-free ds_load_b128.
//   rec16(m, ts, c, lane) ; m: 0=hi,1=lo ; ts = t*KSTEPS+s ; c: chunk 0/1
#define BREC16(m, ts, c, lane) (((((m) * TSTEPS + (ts)) * 2 + (c)) * 32) + (lane))
#define BPACK_BYTES (2 * TSTEPS * 2 * 32 * 16)   // 114,688 B

union BOp { v16bf v; v8bf h[2]; };

// ---------------------------------------------------------------------------
// Pack clusters (f32 [100,256]) into split-bf16 hi/lo for the WMMA B-matrix
// (32x16, 16-bit) in this code's contiguous-K order (same as A): lane l = column n=l&15, K=(l>>4)*16..+15.
// ---------------------------------------------------------------------------
__global__ __launch_bounds__(32)
void pack_clusters_kernel(const float* __restrict__ cl, v8bf* __restrict__ gB)
{
    const int ts   = blockIdx.x;                 // 0..TSTEPS-1
    const int t    = ts / KSTEPS;
    const int s    = ts % KSTEPS;
    const int lane = threadIdx.x;
    const int n    = t * 16 + (lane & 15);       // cluster (may be pad)
    const int k0   = s * 32 + (lane >> 4) * 16;  // first K of this lane record

    v8bf h[2], l[2];
    #pragma unroll
    for (int c = 0; c < 2; ++c) {
        #pragma unroll
        for (int j = 0; j < 8; ++j) {
            float v = (n < KCLUST) ? cl[(long)n * DDIM + k0 + c * 8 + j] : 0.0f;
            __bf16 hh = (__bf16)v;
            h[c][j] = hh;
            l[c][j] = (__bf16)(v - (float)hh);
        }
    }
    gB[BREC16(0, ts, 0, lane)] = h[0];
    gB[BREC16(0, ts, 1, lane)] = h[1];
    gB[BREC16(1, ts, 0, lane)] = l[0];
    gB[BREC16(1, ts, 1, lane)] = l[1];
}

// Exact f32 squared norms of clusters (zero for pad columns).
__global__ __launch_bounds__(32)
void c2_kernel(const float* __restrict__ cl, float* __restrict__ c2)
{
    const int n = blockIdx.x * 32 + threadIdx.x;
    if (n >= NTILES * 16) return;
    float s = 0.0f;
    if (n < KCLUST) {
        #pragma unroll 4
        for (int k = 0; k < DDIM; ++k) {
            float v = cl[(long)n * DDIM + k];
            s = fmaf(v, v, s);
        }
    }
    c2[n] = s;
}

// ---------------------------------------------------------------------------
// Main kernel: wave owns 32 rows x 112 (padded) clusters.
//   z.c ~= zh.ch + zl.ch + zh.cl   (f32 WMMA accumulate, split-bf16)
// B staged once per block in LDS and shared by all 8 waves.
// ---------------------------------------------------------------------------
__global__ __launch_bounds__(256)
void dec_q_kernel(const float* __restrict__ z,
                  const v8bf* __restrict__ gB,
                  const float* __restrict__ c2,
                  float* __restrict__ out)
{
    extern __shared__ char ldsB[];

    // ---- stage packed B (114,688 B) into LDS: straight 16B-record copy ----
    {
        const v4f* src = (const v4f*)gB;
        v4f*       dst = (v4f*)ldsB;
        for (int i = threadIdx.x; i < BPACK_BYTES / 16; i += 256)
            dst[i] = src[i];
    }
    __syncthreads();
    const v8bf* lB = (const v8bf*)ldsB;

    const int lane = threadIdx.x & 31;
    const int wave = threadIdx.x >> 5;
    const int hl   = lane >> 4;
    const int nlo  = lane & 15;
    const long m0  = ((long)blockIdx.x * 8 + wave) * (16 * ROWT);

    v8f acc[ROWT][NTILES];
    #pragma unroll
    for (int r = 0; r < ROWT; ++r)
        #pragma unroll
        for (int t = 0; t < NTILES; ++t) acc[r][t] = (v8f)0.0f;

    // A layout: lane l = row l&15, runs k = s*32 + hl*16 + {0..15}.
    const float* zr[ROWT];
    #pragma unroll
    for (int r = 0; r < ROWT; ++r)
        zr[r] = z + (m0 + r * 16 + nlo) * DDIM + hl * 16;

    float z2p[ROWT];
    #pragma unroll
    for (int r = 0; r < ROWT; ++r) z2p[r] = 0.0f;

    for (int s = 0; s < KSTEPS; ++s) {
        // Prefetch next k-step's z data (global_prefetch_b8, no regs held).
        if (s < KSTEPS - 1) {
            #pragma unroll
            for (int r = 0; r < ROWT; ++r) {
                __builtin_prefetch(zr[r] + (s + 1) * 32, 0, 1);
                __builtin_prefetch(zr[r] + (s + 1) * 32 + 16, 0, 1);
            }
        }

        v16bf ah[ROWT], al[ROWT];
        #pragma unroll
        for (int r = 0; r < ROWT; ++r) {
            v4f x0 = *(const v4f*)(zr[r] + s * 32 + 0);
            v4f x1 = *(const v4f*)(zr[r] + s * 32 + 4);
            v4f x2 = *(const v4f*)(zr[r] + s * 32 + 8);
            v4f x3 = *(const v4f*)(zr[r] + s * 32 + 12);
            float xf[16];
            #pragma unroll
            for (int j = 0; j < 4; ++j) {
                xf[j] = x0[j]; xf[4 + j] = x1[j]; xf[8 + j] = x2[j]; xf[12 + j] = x3[j];
            }
            #pragma unroll
            for (int j = 0; j < 16; ++j) {
                float v = xf[j];
                z2p[r] = fmaf(v, v, z2p[r]);
                __bf16 hh = (__bf16)v;
                ah[r][j] = hh;
                al[r][j] = (__bf16)(v - (float)hh);
            }
        }

        // ---- tile loop, B double-buffered from LDS ----
        BOp bh, bl;
        bh.h[0] = lB[BREC16(0, s, 0, lane)];
        bh.h[1] = lB[BREC16(0, s, 1, lane)];
        bl.h[0] = lB[BREC16(1, s, 0, lane)];
        bl.h[1] = lB[BREC16(1, s, 1, lane)];

        #pragma unroll
        for (int t = 0; t < NTILES; ++t) {
            BOp bhn, bln;
            if (t < NTILES - 1) {
                const int tsn = (t + 1) * KSTEPS + s;
                bhn.h[0] = lB[BREC16(0, tsn, 0, lane)];
                bhn.h[1] = lB[BREC16(0, tsn, 1, lane)];
                bln.h[0] = lB[BREC16(1, tsn, 0, lane)];
                bln.h[1] = lB[BREC16(1, tsn, 1, lane)];
            }
            #pragma unroll
            for (int r = 0; r < ROWT; ++r) {
                acc[r][t] = __builtin_amdgcn_wmma_f32_16x16x32_bf16(
                                false, ah[r], false, bh.v, (short)0, acc[r][t], false, false);
                acc[r][t] = __builtin_amdgcn_wmma_f32_16x16x32_bf16(
                                false, al[r], false, bh.v, (short)0, acc[r][t], false, false);
                acc[r][t] = __builtin_amdgcn_wmma_f32_16x16x32_bf16(
                                false, ah[r], false, bl.v, (short)0, acc[r][t], false, false);
            }
            if (t < NTILES - 1) { bh = bhn; bl = bln; }
        }
    }

    // ---- epilogue: q = 1/(1+max(0, z2+c2-2*dot)), row-normalize ----
    float z2row[ROWT][8];
    #pragma unroll
    for (int r = 0; r < ROWT; ++r) {
        float z2c = z2p[r] + __shfl_xor(z2p[r], 16, 32);
        #pragma unroll
        for (int v = 0; v < 8; ++v) z2row[r][v] = __shfl(z2c, v + 8 * hl, 32);
    }

    float c2n[NTILES];
    #pragma unroll
    for (int t = 0; t < NTILES; ++t) c2n[t] = c2[t * 16 + nlo];

    float rs[ROWT][8];
    #pragma unroll
    for (int r = 0; r < ROWT; ++r)
        #pragma unroll
        for (int v = 0; v < 8; ++v) rs[r][v] = 0.0f;

    #pragma unroll
    for (int t = 0; t < NTILES; ++t) {
        const bool valid = (t * 16 + nlo) < KCLUST;
        #pragma unroll
        for (int r = 0; r < ROWT; ++r) {
            #pragma unroll
            for (int v = 0; v < 8; ++v) {
                float d2 = z2row[r][v] + c2n[t] - 2.0f * acc[r][t][v];
                d2 = fmaxf(d2, 0.0f);
                float q = valid ? (1.0f / (1.0f + d2)) : 0.0f;
                acc[r][t][v] = q;
                rs[r][v] += q;
            }
        }
    }

    #pragma unroll
    for (int r = 0; r < ROWT; ++r) {
        #pragma unroll
        for (int v = 0; v < 8; ++v) {
            float x = rs[r][v];
            x += __shfl_xor(x, 1, 32);
            x += __shfl_xor(x, 2, 32);
            x += __shfl_xor(x, 4, 32);
            x += __shfl_xor(x, 8, 32);
            rs[r][v] = 1.0f / x;
        }
    }

    #pragma unroll
    for (int t = 0; t < NTILES; ++t) {
        const int n = t * 16 + nlo;
        if (n < KCLUST) {
            #pragma unroll
            for (int r = 0; r < ROWT; ++r) {
                #pragma unroll
                for (int v = 0; v < 8; ++v) {
                    const long row = m0 + r * 16 + v + 8 * hl;
                    out[row * KCLUST + n] = acc[r][t][v] * rs[r][v];
                }
            }
        }
    }
}

// ---------------------------------------------------------------------------
extern "C" void kernel_launch(void* const* d_in, const int* in_sizes, int n_in,
                              void* d_out, int out_size, void* d_ws, size_t ws_size,
                              hipStream_t stream)
{
    const float* z  = (const float*)d_in[0];   // [N, 256] f32
    const float* cl = (const float*)d_in[1];   // [100, 256] f32
    float* out      = (float*)d_out;           // [N, 100] f32

    char*  ws = (char*)d_ws;
    v8bf*  gB = (v8bf*)ws;                        // packed split-bf16 B
    float* c2 = (float*)(ws + BPACK_BYTES);       // cluster norms

    const long N = in_sizes[0] / DDIM;            // 131072

    pack_clusters_kernel<<<TSTEPS, 32, 0, stream>>>(cl, gB);
    c2_kernel<<<(NTILES * 16 + 31) / 32, 32, 0, stream>>>(cl, c2);

    // 8 waves/block, 32 rows/wave -> 256 rows per block
    const int blocks = (int)(N / (8 * 16 * ROWT));
    dec_q_kernel<<<blocks, 256, BPACK_BYTES, stream>>>(z, gB, c2, out);
}
